// VectorQuantizer_19215683682406
// MI455X (gfx1250) — matrix operand order repaired and checked by reading, not compiled, not run on hardware
//
#include <hip/hip_runtime.h>

// ---------------- CDNA5 vector types ----------------
typedef __attribute__((ext_vector_type(4)))  __bf16 v4bf;
typedef __attribute__((ext_vector_type(8)))  __bf16 v8bf;
typedef __attribute__((ext_vector_type(16))) __bf16 v16bf;
typedef __attribute__((ext_vector_type(8)))  float  v8f;

#ifndef __has_builtin
#define __has_builtin(x) 0
#endif
#if __has_builtin(__builtin_amdgcn_global_load_async_to_lds_b128)
#define ASYNC_LDS 1
#else
#define ASYNC_LDS 0
#endif

#define D        256
#define KEMB     4096
#define NROWS    65536            // 16*4096
#define MROWS    128              // rows per block (16 per wave, 8 waves)
#define NCH      32               // codebook columns per staged chunk
#define NCHUNKS  (KEMB / NCH)     // 128
#define AST      264              // padded LDS row stride (elements): 528 B, conflict-free b128
#define QELEMS   16777216         // 16*4096*256
#define NBLK     (NROWS / MROWS)  // 512

#if ASYNC_LDS
// Exact parameter element type from hipcc's diagnostic: vector_size(16) int
typedef int v4i_vs __attribute__((__vector_size__(4 * sizeof(int))));
typedef __attribute__((address_space(1))) v4i_vs* gasync_t;
typedef __attribute__((address_space(3))) v4i_vs* lasync_t;
#endif

__device__ __forceinline__ void waitAsync() {
#if ASYNC_LDS
#if __has_builtin(__builtin_amdgcn_s_wait_asynccnt)
    __builtin_amdgcn_s_wait_asynccnt(0);
#else
    asm volatile("s_wait_asynccnt 0x0" ::: "memory");
#endif
#endif
}

// Stage one 32x256 bf16 codebook chunk into a padded LDS buffer.
// 256 threads: 8 threads/row, 32 bf16 (4 x b128) each.
__device__ __forceinline__ void stageB(const __bf16* __restrict__ embB,
                                       __bf16* dstBuf, int ch, int t) {
    const int row = t >> 3;
    const int d0  = (t & 7) * 32;
    const __bf16* gsrc = embB + (size_t)(ch * NCH + row) * D + d0;
    __bf16* ldst = dstBuf + row * AST + d0;
#if ASYNC_LDS
    gasync_t g = (gasync_t)(const char*)gsrc;
    lasync_t l = (lasync_t)(char*)ldst;
    __builtin_amdgcn_global_load_async_to_lds_b128(g, l, 0,  0);
    __builtin_amdgcn_global_load_async_to_lds_b128(g, l, 16, 0);
    __builtin_amdgcn_global_load_async_to_lds_b128(g, l, 32, 0);
    __builtin_amdgcn_global_load_async_to_lds_b128(g, l, 48, 0);
#else
    const v8bf* src = (const v8bf*)gsrc;
    #pragma unroll
    for (int j = 0; j < 4; j++) *(v8bf*)(ldst + j * 8) = src[j];
#endif
}

// ------------------------------------------------------------------
// Pass 0: codebook fp32 -> bf16, and ||e||^2 per code
// ------------------------------------------------------------------
__global__ __launch_bounds__(256) void vq_prep(const float* __restrict__ emb,
                                               unsigned short* __restrict__ embB_,
                                               float* __restrict__ embSq) {
    __bf16* embB = (__bf16*)embB_;
    const int row = blockIdx.x;
    const int t   = threadIdx.x;
    float e = emb[(size_t)row * D + t];
    embB[(size_t)row * D + t] = (__bf16)e;
    float s = e * e;
    #pragma unroll
    for (int m = 16; m >= 1; m >>= 1) s += __shfl_xor(s, m, 32);
    __shared__ float wsum[8];
    if ((t & 31) == 0) wsum[t >> 5] = s;
    __syncthreads();
    if (t == 0) {
        float tot = 0.f;
        #pragma unroll
        for (int i = 0; i < 8; i++) tot += wsum[i];
        embSq[row] = tot;
    }
}

// ------------------------------------------------------------------
// Pass 1: distances via WMMA bf16, running argmin, gather + loss partial
// ------------------------------------------------------------------
__global__ __launch_bounds__(256) void vq_main(const float* __restrict__ x,
                                               const float* __restrict__ embF,
                                               const unsigned short* __restrict__ embB_,
                                               const float* __restrict__ embSq,
                                               float* __restrict__ outQ,
                                               float* __restrict__ outIdx,
                                               float* __restrict__ partial) {
    const __bf16* embB = (const __bf16*)embB_;

    __shared__ __bf16 As[MROWS * AST];      // 128 x 256 bf16, padded  (~67.6 KB)
    __shared__ __bf16 Bs[2][NCH * AST];     // ping-pong 32 x 256 bf16 (~2x16.9 KB)
    __shared__ float  sqAll[KEMB];          // all ||e||^2             (16 KB)
    __shared__ int    idxS[MROWS];
    __shared__ float  wsum[8];

    const int t    = threadIdx.x;
    const int lane = t & 31;
    const int wv   = t >> 5;          // 8 waves, one 16-row M-group each
    const int hi   = lane >> 4;       // lane half
    const int lm   = lane & 15;
    const int r0   = blockIdx.x * MROWS;

    // ---- stage input tile: fp32 global -> bf16 LDS (padded) ----
    {
        const int row = t >> 1;
        const int d0  = (t & 1) * 128;
        const float4* src = (const float4*)(x + (size_t)(r0 + row) * D + d0);
        __bf16* dst = &As[row * AST + d0];
        #pragma unroll
        for (int j = 0; j < 32; j++) {
            float4 f = src[j];
            v4bf p;
            p[0] = (__bf16)f.x; p[1] = (__bf16)f.y;
            p[2] = (__bf16)f.z; p[3] = (__bf16)f.w;
            *(v4bf*)(dst + j * 4) = p;
        }
    }
    // ---- stage all ||e||^2 into LDS (one time) ----
    {
        const float4* s4 = (const float4*)embSq;
        float4* d4 = (float4*)sqAll;
        #pragma unroll
        for (int j = 0; j < 4; j++) d4[t + j * 256] = s4[t + j * 256];
    }
    __syncthreads();

    // ---- hoist A fragments (full D=256) into registers: ISA 16-bit A layout ----
    // lane half 0: K = kc*32 + {0..7, 16..23}; half 1: K = kc*32 + {8..15, 24..31}
    v16bf aF[8];
    {
        const __bf16* abase = &As[(wv * 16 + lm) * AST];
        #pragma unroll
        for (int kc = 0; kc < 8; kc++) {
            const int kb = kc * 32 + hi * 8;
            v8bf lo = *(const v8bf*)(abase + kb);
            v8bf hh = *(const v8bf*)(abase + kb + 16);
            aF[kc] = __builtin_shufflevector(lo, hh, 0,1,2,3,4,5,6,7,8,9,10,11,12,13,14,15);
        }
    }

    float bestD[8];
    int   bestI[8];
    #pragma unroll
    for (int v = 0; v < 8; v++) { bestD[v] = 3.4e38f; bestI[v] = 0; }

    // ---- prologue: stage chunk 0 ----
    stageB(embB, Bs[0], 0, t);

    for (int ch = 0; ch < NCHUNKS; ch++) {
        waitAsync();                     // my async writes for Bs[ch&1] complete
        __syncthreads();                 // everyone's writes visible; prev reads drained
        if (ch + 1 < NCHUNKS)            // overlap next chunk's DMA with compute
            stageB(embB, Bs[(ch + 1) & 1], ch + 1, t);

        // ---- 16x32 slab per wave: two 16x16 tiles, K=256 via 8 WMMAs each ----
        v8f c0 = {};
        v8f c1 = {};
        const __bf16* bbase0 = &Bs[ch & 1][lm * AST];          // cols 0..15 of chunk
        const __bf16* bbase1 = &Bs[ch & 1][(16 + lm) * AST];   // cols 16..31 of chunk
        #pragma unroll
        for (int kc = 0; kc < 8; kc++) {
            const int kb = kc * 32 + hi * 8;
            v8bf lo0 = *(const v8bf*)(bbase0 + kb);
            v8bf hh0 = *(const v8bf*)(bbase0 + kb + 16);
            v8bf lo1 = *(const v8bf*)(bbase1 + kb);
            v8bf hh1 = *(const v8bf*)(bbase1 + kb + 16);
            v16bf b0 = __builtin_shufflevector(lo0, hh0, 0,1,2,3,4,5,6,7,8,9,10,11,12,13,14,15);
            v16bf b1 = __builtin_shufflevector(lo1, hh1, 0,1,2,3,4,5,6,7,8,9,10,11,12,13,14,15);
            c0 = __builtin_amdgcn_wmma_f32_16x16x32_bf16(
                     false, aF[kc], false, b0, (short)0, c0, false, false);
            c1 = __builtin_amdgcn_wmma_f32_16x16x32_bf16(
                     false, aF[kc], false, b1, (short)0, c1, false, false);
        }

        // ---- distance = ||e||^2 - 2 x.e ; running argmin (cols lm, then 16+lm) ----
        const int n0 = ch * NCH + lm;
        const int n1 = n0 + 16;
        const float es0 = sqAll[n0];
        const float es1 = sqAll[n1];
        #pragma unroll
        for (int v = 0; v < 8; v++) {
            float d0_ = es0 - 2.0f * c0[v];
            if (d0_ < bestD[v]) { bestD[v] = d0_; bestI[v] = n0; }
            float d1_ = es1 - 2.0f * c1[v];
            if (d1_ < bestD[v]) { bestD[v] = d1_; bestI[v] = n1; }
        }
    }

    // ---- butterfly argmin across the 16 lanes sharing each row M ----
    // (each lane saw cols == lm (mod 16); union over 16 lanes = all 4096 cols)
    #pragma unroll
    for (int v = 0; v < 8; v++) {
        float d = bestD[v]; int i = bestI[v];
        #pragma unroll
        for (int m = 1; m < 16; m <<= 1) {
            float d2 = __shfl_xor(d, m, 32);
            int   i2 = __shfl_xor(i, m, 32);
            if (d2 < d || (d2 == d && i2 < i)) { d = d2; i = i2; }
        }
        bestD[v] = d; bestI[v] = i;
    }
    if (lm == 0) {
        #pragma unroll
        for (int v = 0; v < 8; v++) {
            const int m = v + 8 * hi;           // C layout: M = v + 8*(lane/16)
            idxS[wv * 16 + m] = bestI[v];
            outIdx[r0 + wv * 16 + m] = (float)bestI[v];
        }
    }
    __syncthreads();

    // ---- gather quantized (fp32) + loss partial ----
    float ls = 0.f;
    {
        const int row = t >> 1;
        const int d0  = (t & 1) * 128;
        const int idx = idxS[row];
        const float4* q   = (const float4*)(embF + (size_t)idx * D + d0);
        const float4* xin = (const float4*)(x + (size_t)(r0 + row) * D + d0);
        float4* o = (float4*)(outQ + (size_t)(r0 + row) * D + d0);
        #pragma unroll
        for (int j = 0; j < 32; j++) {
            float4 qq = q[j], xx = xin[j];
            o[j] = qq;
            float a0 = qq.x - xx.x, a1 = qq.y - xx.y;
            float a2 = qq.z - xx.z, a3 = qq.w - xx.w;
            ls += a0 * a0 + a1 * a1 + a2 * a2 + a3 * a3;
        }
    }
    #pragma unroll
    for (int m = 16; m >= 1; m >>= 1) ls += __shfl_xor(ls, m, 32);
    if (lane == 0) wsum[wv] = ls;
    __syncthreads();
    if (t == 0) {
        float tot = 0.f;
        #pragma unroll
        for (int i = 0; i < 8; i++) tot += wsum[i];
        partial[blockIdx.x] = tot;
    }
}

// ------------------------------------------------------------------
// Pass 2: reduce 512 partials -> mean loss
// ------------------------------------------------------------------
__global__ __launch_bounds__(256) void vq_finalize(const float* __restrict__ partial,
                                                   float* __restrict__ outLoss) {
    const int t = threadIdx.x;
    float s = partial[t] + partial[t + 256];
    #pragma unroll
    for (int m = 16; m >= 1; m >>= 1) s += __shfl_xor(s, m, 32);
    __shared__ float wsum[8];
    if ((t & 31) == 0) wsum[t >> 5] = s;
    __syncthreads();
    if (t == 0) {
        float tot = 0.f;
        #pragma unroll
        for (int i = 0; i < 8; i++) tot += wsum[i];
        *outLoss = tot / (float)QELEMS;
    }
}

// ------------------------------------------------------------------
extern "C" void kernel_launch(void* const* d_in, const int* in_sizes, int n_in,
                              void* d_out, int out_size, void* d_ws, size_t ws_size,
                              hipStream_t stream) {
    const float* x   = (const float*)d_in[0];   // [16,4096,256] fp32
    const float* emb = (const float*)d_in[1];   // [4096,256]    fp32

    float* out     = (float*)d_out;
    float* outQ    = out;                 // 16777216 floats
    float* outLoss = out + QELEMS;        // 1 float
    float* outIdx  = out + QELEMS + 1;    // 65536 floats (indices as float)

    char* ws = (char*)d_ws;
    unsigned short* embB  = (unsigned short*)ws;                          // 2 MB bf16 codebook
    float*          embSq = (float*)(ws + (size_t)KEMB * D * 2);          // 16 KB
    float*          part  = (float*)(ws + (size_t)KEMB * D * 2 + KEMB*4); // 2 KB

    vq_prep<<<KEMB, 256, 0, stream>>>(emb, embB, embSq);
    vq_main<<<NBLK, 256, 0, stream>>>(x, emb, embB, embSq, outQ, outIdx, part);
    vq_finalize<<<1, 256, 0, stream>>>(part, outLoss);
}
